// QuantSiglipEncoder_35003983463156
// MI455X (gfx1250) — 16-bit and 8-bit matrix operand order repaired and checked by reading, compiled, not run on hardware
//
#include <hip/hip_runtime.h>
#include <cstdint>
#include <cstddef>

typedef __attribute__((ext_vector_type(8)))  int       v8i;
typedef __attribute__((ext_vector_type(8)))  float     v8f;
typedef __attribute__((ext_vector_type(16))) _Float16  v16h;

#define LNUM   6
#define DMODEL 1152
#define NHEAD  16
#define HDIM   72
#define FFDIM  4304
#define FFPAD  4352
#define BATCH  16
#define SEQ    1024
#define NTOK   (BATCH*SEQ)
#define QKVN   (3*DMODEL)

__device__ __forceinline__ float wred_sum(float v) {
  for (int o = 16; o; o >>= 1) v += __shfl_xor(v, o, 32);
  return v;
}
__device__ __forceinline__ float wred_max(float v) {
  for (int o = 16; o; o >>= 1) v = fmaxf(v, __shfl_xor(v, o, 32));
  return v;
}
__device__ __forceinline__ float gelu_tanh(float x) {
  float x3 = x * x * x;
  return 0.5f * x * (1.f + tanhf(0.7978845608028654f * (x + 0.044715f * x3)));
}

__global__ __launch_bounds__(256) void convert_w_kernel(
    const float* __restrict__ w, int8_t* __restrict__ out,
    int O, int K, int Opad, int Kpad)
{
  long long idx = (long long)blockIdx.x * 256 + threadIdx.x;
  long long total = (long long)LNUM * Kpad * Opad;
  if (idx >= total) return;
  int op = (int)(idx % Opad);
  int kp = (int)((idx / Opad) % Kpad);
  int l  = (int)(idx / ((long long)Opad * Kpad));
  int v = 0;
  if (op < O && kp < K)
    v = __float2int_rn(w[((long long)l * O + op) * K + kp]);
  v = v < -127 ? -127 : (v > 127 ? 127 : v);
  out[idx] = (int8_t)v;
}

__global__ __launch_bounds__(256) void ln_quant_kernel(
    const float* __restrict__ x, const float* __restrict__ g, const float* __restrict__ b,
    int8_t* __restrict__ q, float* __restrict__ s)
{
  int t = blockIdx.x;
  const float* row = x + (size_t)t * DMODEL;
  int lane = threadIdx.x & 31, wv = threadIdx.x >> 5;
  __shared__ float t0[8], t1[8];
  float sum = 0.f, sq = 0.f;
  for (int i = threadIdx.x; i < DMODEL; i += 256) { float v = row[i]; sum += v; sq += v * v; }
  sum = wred_sum(sum); sq = wred_sum(sq);
  if (!lane) { t0[wv] = sum; t1[wv] = sq; }
  __syncthreads();
  float S = 0.f, Q = 0.f;
  for (int i = 0; i < 8; ++i) { S += t0[i]; Q += t1[i]; }
  float mean = S * (1.f / DMODEL);
  float var  = Q * (1.f / DMODEL) - mean * mean;
  float rstd = rsqrtf(var + 1e-6f);
  __syncthreads();
  float mx = 0.f;
  for (int i = threadIdx.x; i < DMODEL; i += 256) {
    float v = (row[i] - mean) * rstd * g[i] + b[i];
    mx = fmaxf(mx, fabsf(v));
  }
  mx = wred_max(mx);
  if (!lane) t0[wv] = mx;
  __syncthreads();
  float MX = 0.f;
  for (int i = 0; i < 8; ++i) MX = fmaxf(MX, t0[i]);
  if (threadIdx.x == 0) s[t] = MX * (1.f / 127.f);
  float inv = MX > 0.f ? 127.f / MX : 0.f;
  for (int i = threadIdx.x; i < DMODEL; i += 256) {
    float v = (row[i] - mean) * rstd * g[i] + b[i];
    float qv = fminf(fmaxf(rintf(v * inv), -127.f), 127.f);
    q[(size_t)t * FFPAD + i] = (int8_t)qv;
  }
}

__global__ __launch_bounds__(256) void quant_kernel(
    const float* __restrict__ in, int istride, int W,
    int8_t* __restrict__ q, int qstride, float* __restrict__ s)
{
  int t = blockIdx.x;
  const float* row = in + (size_t)t * istride;
  int lane = threadIdx.x & 31, wv = threadIdx.x >> 5;
  __shared__ float t0[8];
  float mx = 0.f;
  for (int i = threadIdx.x; i < W; i += 256) mx = fmaxf(mx, fabsf(row[i]));
  mx = wred_max(mx);
  if (!lane) t0[wv] = mx;
  __syncthreads();
  float MX = 0.f;
  for (int i = 0; i < 8; ++i) MX = fmaxf(MX, t0[i]);
  if (threadIdx.x == 0) s[t] = MX * (1.f / 127.f);
  float inv = MX > 0.f ? 127.f / MX : 0.f;
  for (int i = threadIdx.x; i < W; i += 256) {
    float qv = fminf(fmaxf(rintf(row[i] * inv), -127.f), 127.f);
    q[(size_t)t * qstride + i] = (int8_t)qv;
  }
}

__global__ __launch_bounds__(256) void gelu_quant_kernel(
    const float* __restrict__ hin, int istride, int W,
    int8_t* __restrict__ q, int qstride, float* __restrict__ s)
{
  int t = blockIdx.x;
  const float* row = hin + (size_t)t * istride;
  int lane = threadIdx.x & 31, wv = threadIdx.x >> 5;
  __shared__ float t0[8];
  float mx = 0.f;
  for (int i = threadIdx.x; i < W; i += 256) mx = fmaxf(mx, fabsf(gelu_tanh(row[i])));
  mx = wred_max(mx);
  if (!lane) t0[wv] = mx;
  __syncthreads();
  float MX = 0.f;
  for (int i = 0; i < 8; ++i) MX = fmaxf(MX, t0[i]);
  if (threadIdx.x == 0) s[t] = MX * (1.f / 127.f);
  float inv = MX > 0.f ? 127.f / MX : 0.f;
  for (int i = threadIdx.x; i < qstride; i += 256) {
    int8_t qv = 0;
    if (i < W) {
      float a = gelu_tanh(row[i]);
      qv = (int8_t)fminf(fmaxf(rintf(a * inv), -127.f), 127.f);
    }
    q[(size_t)t * qstride + i] = qv;
  }
}

#define TILE_BYTES (128*64 + 64*64)

__global__ __launch_bounds__(256) void gemm_w8a8_kernel(
    const int8_t* __restrict__ A, int lda,
    const int8_t* __restrict__ B, int ldb,
    const float* __restrict__ sx, const float* __restrict__ ws,
    const float* __restrict__ bias, const float* __restrict__ res,
    float* __restrict__ C, int ldc, int Nreal, int K)
{
  __shared__ int8_t smem[2 * TILE_BYTES];
  const int tid = threadIdx.x, lane = tid & 31, wv = tid >> 5;
  const int hf = lane >> 4, r = lane & 15;
  const int m0 = blockIdx.y * 128, n0 = blockIdx.x * 64;
  const int wm = (wv & 3) * 32, wn = (wv >> 2) * 32;
  const int brow = tid >> 2, bcol = (tid & 3) << 4;
  v8i acc[2][2] = {};

  auto issue_tile = [&](int k0, int bufsel) {
    int8_t* tb = smem + bufsel * TILE_BYTES;
#pragma unroll
    for (int i = 0; i < 2; ++i) {
      int idx = tid + i * 256;
      int arow = idx >> 2, acol = (idx & 3) << 4;
      unsigned lo = (unsigned)(uintptr_t)(tb + arow * 64 + acol);
      unsigned long long ga =
          (unsigned long long)(A + (size_t)(m0 + arow) * lda + k0 + acol);
      asm volatile("global_load_async_to_lds_b128 %0, %1, off"
                   :: "v"(lo), "v"(ga) : "memory");
    }
    unsigned lo = (unsigned)(uintptr_t)(tb + 8192 + brow * 64 + bcol);
    unsigned long long gb =
        (unsigned long long)(B + (size_t)(k0 + brow) * ldb + n0 + bcol);
    asm volatile("global_load_async_to_lds_b128 %0, %1, off"
                 :: "v"(lo), "v"(gb) : "memory");
  };

  issue_tile(0, 0);
  asm volatile("s_wait_asynccnt 0x0" ::: "memory");
  __syncthreads();

  int buf = 0;
  for (int k0 = 0; k0 < K; k0 += 64) {
    if (k0 + 64 < K) issue_tile(k0 + 64, buf ^ 1);

    const int8_t* sA = smem + buf * TILE_BYTES;
    const int8_t* sB = sA + 8192;

    v8i af[2], bf[2];
#pragma unroll
    for (int mi = 0; mi < 2; ++mi) {
      const int8_t* base = sA + (wm + mi * 16 + r) * 64 + hf * 8;
      int2 p0 = *(const int2*)(base + 0);
      int2 p1 = *(const int2*)(base + 16);
      int2 p2 = *(const int2*)(base + 32);
      int2 p3 = *(const int2*)(base + 48);
      v8i a; a[0]=p0.x; a[1]=p0.y; a[2]=p1.x; a[3]=p1.y;
             a[4]=p2.x; a[5]=p2.y; a[6]=p3.x; a[7]=p3.y;
      af[mi] = a;
    }
#pragma unroll
    for (int ni = 0; ni < 2; ++ni) {
      const int8_t* base = sB + wn + ni * 16 + r;
      v8i bm;
      for (int j = 0; j < 8; ++j) {
        unsigned w = 0;
        for (int bb = 0; bb < 4; ++bb) {
          int s = 4 * j + bb;
          int kk = (s & 7) + 8 * hf + 16 * (s >> 3);
          w |= (unsigned)(uint8_t)base[kk * 64] << (8 * bb);
        }
        bm[j] = (int)w;
      }
      bf[ni] = bm;
    }
#pragma unroll
    for (int mi = 0; mi < 2; ++mi)
#pragma unroll
      for (int ni = 0; ni < 2; ++ni)
        acc[mi][ni] = __builtin_amdgcn_wmma_i32_16x16x64_iu8(
            true, af[mi], true, bf[ni], acc[mi][ni], false, false);

    asm volatile("s_wait_asynccnt 0x0" ::: "memory");
    __syncthreads();
    buf ^= 1;
  }

#pragma unroll
  for (int ni = 0; ni < 2; ++ni) {
    int n = n0 + wn + ni * 16 + r;
    if (n >= Nreal) continue;
    float wsn = ws[n], bn = bias[n];
#pragma unroll
    for (int mi = 0; mi < 2; ++mi) {
#pragma unroll
      for (int e = 0; e < 8; ++e) {
        int m = m0 + wm + mi * 16 + e + 8 * hf;
        float v = (float)acc[mi][ni][e] * sx[m] * wsn + bn;
        if (res) v += res[(size_t)m * ldc + n];
        C[(size_t)m * ldc + n] = v;
      }
    }
  }
}

__global__ __launch_bounds__(256) void attention_kernel(
    const float* __restrict__ qkv, float* __restrict__ o)
{
  const int mblk = blockIdx.x, hh = blockIdx.y, bb = blockIdx.z;
  const int tid = threadIdx.x, lane = tid & 31, wv = tid >> 5;
  const int hf = lane >> 4, r = lane & 15;

  __shared__ _Float16 sQ[128][96];
  __shared__ _Float16 sKT[96][32];
  __shared__ _Float16 sV[32][80];
  __shared__ float    sS[8][16][32];
  __shared__ float    sRow[8][16];

  const float scale = 0.1178511301977579f;
  const size_t tok0 = (size_t)bb * SEQ;
  const int cbase = hh * HDIM;

  for (int idx = tid; idx < 128 * 96; idx += 256) {
    int row = idx / 96, c = idx % 96;
    float v = (c < HDIM)
        ? qkv[(tok0 + mblk * 128 + row) * QKVN + cbase + c] * scale : 0.f;
    sQ[row][c] = (_Float16)v;
  }
  __syncthreads();

  v16h qf[3];
#pragma unroll
  for (int c = 0; c < 3; ++c) {
    const char* base = (const char*)&sQ[wv * 16 + r][c * 32];
    union { int4 i4[2]; v16h h16; } u;
    u.i4[0] = *(const int4*)(base + hf * 16);
    u.i4[1] = *(const int4*)(base + 32 + hf * 16);
    qf[c] = u.h16;
  }

  v8f oacc[5] = {};
  float rowm = -1e30f, rowl = 0.f;

  for (int kt = 0; kt < 32; ++kt) {
    __syncthreads();
    for (int idx = tid; idx < 32 * 96; idx += 256) {
      int row = idx / 96, c = idx % 96;
      float v = (c < HDIM)
          ? qkv[(tok0 + kt * 32 + row) * QKVN + DMODEL + cbase + c] : 0.f;
      sKT[c][row] = (_Float16)v;
    }
    for (int idx = tid; idx < 32 * 80; idx += 256) {
      int row = idx / 80, c = idx % 80;
      float v = (c < HDIM)
          ? qkv[(tok0 + kt * 32 + row) * QKVN + 2 * DMODEL + cbase + c] : 0.f;
      sV[row][c] = (_Float16)v;
    }
    __syncthreads();

    v8f sacc[2] = {};
#pragma unroll
    for (int ni = 0; ni < 2; ++ni) {
#pragma unroll
      for (int c = 0; c < 3; ++c) {
        union { int4 i4[2]; v16h h16; } u;
        for (int s = 0; s < 16; ++s) {
          int kk = (s & 7) + 8 * hf + 16 * (s >> 3);
          u.h16[s] = sKT[c * 32 + kk][ni * 16 + r];
        }
        sacc[ni] = __builtin_amdgcn_wmma_f32_16x16x32_f16(
            false, qf[c], false, u.h16, (short)0, sacc[ni], false, false);
      }
    }

#pragma unroll
    for (int ni = 0; ni < 2; ++ni)
#pragma unroll
      for (int e = 0; e < 8; ++e)
        sS[wv][e + 8 * hf][ni * 16 + r] = sacc[ni][e];
    asm volatile("s_wait_dscnt 0" ::: "memory");

    float mnew = rowm;
    for (int j = 0; j < 32; ++j) mnew = fmaxf(mnew, sS[wv][r][j]);
    float corr = __expf(rowm - mnew);
    float psum = 0.f;
    for (int j = 0; j < 32; ++j) {
      float p = __expf(sS[wv][r][j] - mnew);
      sS[wv][r][j] = p;
      psum += p;
    }
    rowl = rowl * corr + psum;
    rowm = mnew;
    sRow[wv][r] = corr;
    asm volatile("s_wait_dscnt 0" ::: "memory");

    float cf[8];
#pragma unroll
    for (int e = 0; e < 8; ++e) cf[e] = sRow[wv][e + 8 * hf];
#pragma unroll
    for (int ni = 0; ni < 5; ++ni)
#pragma unroll
      for (int e = 0; e < 8; ++e) oacc[ni][e] *= cf[e];

    union { _Float16 ph[16]; v16h pv; } up;
#pragma unroll
    for (int v = 0; v < 8; ++v) {
      int kk = (v & 3) * 2 + (v >> 2) * 16 + hf * 8;
      up.ph[2 * v]     = (_Float16)sS[wv][r][kk];
      up.ph[2 * v + 1] = (_Float16)sS[wv][r][kk + 1];
    }

#pragma unroll
    for (int ni = 0; ni < 5; ++ni) {
      union { int4 i4[2]; v16h h16; } uv;
      for (int s = 0; s < 16; ++s) {
        int kk = (s & 7) + 8 * hf + 16 * (s >> 3);
        uv.h16[s] = sV[kk][ni * 16 + r];
      }
      oacc[ni] = __builtin_amdgcn_wmma_f32_16x16x32_f16(
          false, up.pv, false, uv.h16, (short)0, oacc[ni], false, false);
    }
  }

  sRow[wv][r] = (rowl > 0.f) ? 1.f / rowl : 0.f;
  asm volatile("s_wait_dscnt 0" ::: "memory");
  float inv[8];
#pragma unroll
  for (int e = 0; e < 8; ++e) inv[e] = sRow[wv][e + 8 * hf];
#pragma unroll
  for (int ni = 0; ni < 5; ++ni) {
    int hd = ni * 16 + r;
    if (hd >= HDIM) continue;
#pragma unroll
    for (int e = 0; e < 8; ++e) {
      size_t m = tok0 + mblk * 128 + wv * 16 + e + 8 * hf;
      o[m * DMODEL + cbase + hd] = oacc[ni][e] * inv[e];
    }
  }
}

extern "C" void kernel_launch(void* const* d_in, const int* in_sizes, int n_in,
                              void* d_out, int out_size, void* d_ws, size_t ws_size,
                              hipStream_t stream)
{
  const float* inputs = (const float*)d_in[0];
  const float* ln1_g  = (const float*)d_in[1];
  const float* ln1_b  = (const float*)d_in[2];
  const float* wqkv_q = (const float*)d_in[3];
  const float* wqkv_s = (const float*)d_in[4];
  const float* bqkv   = (const float*)d_in[5];
  const float* wo_q   = (const float*)d_in[6];
  const float* wo_s   = (const float*)d_in[7];
  const float* bo     = (const float*)d_in[8];
  const float* ln2_g  = (const float*)d_in[9];
  const float* ln2_b  = (const float*)d_in[10];
  const float* w1_q   = (const float*)d_in[11];
  const float* w1_s   = (const float*)d_in[12];
  const float* b1     = (const float*)d_in[13];
  const float* w2_q   = (const float*)d_in[14];
  const float* w2_s   = (const float*)d_in[15];
  const float* b2     = (const float*)d_in[16];

  char* ws = (char*)d_ws;
  size_t off = 0;
  auto alloc = [&](size_t bytes) {
    char* p = ws + off;
    off = (off + bytes + 255) & ~(size_t)255;
    return p;
  };
  int8_t* wqkv_i8 = (int8_t*)alloc((size_t)LNUM * 1152 * QKVN);
  int8_t* wo_i8   = (int8_t*)alloc((size_t)LNUM * 1152 * 1152);
  int8_t* w1_i8   = (int8_t*)alloc((size_t)LNUM * 1152 * FFPAD);
  int8_t* w2_i8   = (int8_t*)alloc((size_t)LNUM * FFPAD * 1152);
  int8_t* qx      = (int8_t*)alloc((size_t)NTOK * FFPAD);
  float*  sxb     = (float*)alloc((size_t)NTOK * 4);
  float*  xbuf    = (float*)alloc((size_t)NTOK * DMODEL * 4);
  float*  qkvbuf  = (float*)alloc((size_t)NTOK * QKVN * 4);
  float*  hbuf    = (float*)alloc((size_t)NTOK * FFPAD * 4);
  float*  obuf    = hbuf;

  auto cgrid = [](long long total) { return (unsigned)((total + 255) / 256); };
  convert_w_kernel<<<cgrid((long long)LNUM * 1152 * QKVN), 256, 0, stream>>>(
      wqkv_q, wqkv_i8, QKVN, 1152, QKVN, 1152);
  convert_w_kernel<<<cgrid((long long)LNUM * 1152 * 1152), 256, 0, stream>>>(
      wo_q, wo_i8, 1152, 1152, 1152, 1152);
  convert_w_kernel<<<cgrid((long long)LNUM * 1152 * FFPAD), 256, 0, stream>>>(
      w1_q, w1_i8, FFDIM, 1152, FFPAD, 1152);
  convert_w_kernel<<<cgrid((long long)LNUM * FFPAD * 1152), 256, 0, stream>>>(
      w2_q, w2_i8, 1152, FFDIM, 1152, FFPAD);

  hipMemcpyAsync(xbuf, inputs, (size_t)NTOK * DMODEL * 4,
                 hipMemcpyDeviceToDevice, stream);

  for (int l = 0; l < LNUM; ++l) {
    ln_quant_kernel<<<NTOK, 256, 0, stream>>>(
        xbuf, ln1_g + l * DMODEL, ln1_b + l * DMODEL, qx, sxb);
    gemm_w8a8_kernel<<<dim3(QKVN / 64, NTOK / 128), 256, 0, stream>>>(
        qx, FFPAD, wqkv_i8 + (size_t)l * 1152 * QKVN, QKVN,
        sxb, wqkv_s + (size_t)l * QKVN, bqkv + (size_t)l * QKVN, nullptr,
        qkvbuf, QKVN, QKVN, 1152);
    attention_kernel<<<dim3(SEQ / 128, NHEAD, BATCH), 256, 0, stream>>>(qkvbuf, obuf);
    quant_kernel<<<NTOK, 256, 0, stream>>>(obuf, DMODEL, DMODEL, qx, FFPAD, sxb);
    gemm_w8a8_kernel<<<dim3(1152 / 64, NTOK / 128), 256, 0, stream>>>(
        qx, FFPAD, wo_i8 + (size_t)l * 1152 * 1152, 1152,
        sxb, wo_s + (size_t)l * 1152, bo + (size_t)l * 1152, xbuf,
        xbuf, DMODEL, 1152, 1152);
    ln_quant_kernel<<<NTOK, 256, 0, stream>>>(
        xbuf, ln2_g + l * DMODEL, ln2_b + l * DMODEL, qx, sxb);
    gemm_w8a8_kernel<<<dim3(FFPAD / 64, NTOK / 128), 256, 0, stream>>>(
        qx, FFPAD, w1_i8 + (size_t)l * 1152 * FFPAD, FFPAD,
        sxb, w1_s + (size_t)l * FFDIM, b1 + (size_t)l * FFDIM, nullptr,
        hbuf, FFPAD, FFDIM, 1152);
    gelu_quant_kernel<<<NTOK, 256, 0, stream>>>(hbuf, FFPAD, FFDIM, qx, FFPAD, sxb);
    gemm_w8a8_kernel<<<dim3(1152 / 64, NTOK / 128), 256, 0, stream>>>(
        qx, FFPAD, w2_i8 + (size_t)l * FFPAD * 1152, 1152,
        sxb, w2_s + (size_t)l * 1152, b2 + (size_t)l * 1152, xbuf,
        xbuf, DMODEL, 1152, FFPAD);
  }

  hipMemcpyAsync(d_out, xbuf, (size_t)NTOK * DMODEL * 4,
                 hipMemcpyDeviceToDevice, stream);
  (void)in_sizes; (void)n_in; (void)out_size; (void)ws_size;
}
